// LanguageModel_5909875000161
// MI455X (gfx1250) — compile-verified, matrix operand order checked by reading, not run on hardware
//
#include <hip/hip_runtime.h>
#include <hip/hip_bf16.h>
#include <math.h>

// ---------------- problem constants (match reference) ----------------
constexpr int Bz  = 8;
constexpr int Sz  = 512;
constexpr int Hz  = 768;
constexpr int NHz = 12;
constexpr int DHz = 64;
constexpr int FFz = 3072;
constexpr int Vz  = 21128;
constexpr int Lz  = 12;
constexpr int SEPz = 102;
constexpr int Mtok = Bz * Sz;            // 4096 token rows

// ---------------- vector types ----------------
typedef __bf16 bf16;
typedef __attribute__((ext_vector_type(8)))  __bf16 v8bf;
typedef __attribute__((ext_vector_type(16))) __bf16 v16bf;
typedef __attribute__((ext_vector_type(8)))  float  v8f;

// ---------------- CDNA5 async global->LDS copy (ASYNCcnt-tracked) ----------
// LDS destination address is taken from a VGPR; low 32 bits of a generic LDS pointer are the
// LDS offset on AMDGPU, so truncation gives the right VGPR value.
__device__ __forceinline__ void async_b128(unsigned lds, const void* gaddr) {
  asm volatile("global_load_async_to_lds_b128 %0, %1, off"
               :: "v"(lds), "v"(gaddr) : "memory");
}
__device__ __forceinline__ void wait_async0() {
  asm volatile("s_wait_asynccnt 0x0" ::: "memory");
}

// Fragment load: per the CDNA5 16-bit 16x32 operand layout, each lane needs
// K = {half*8 .. half*8+7} and K = {16+half*8 ..} -> two contiguous 16B LDS reads.
__device__ __forceinline__ v16bf frag_ld(const bf16* p) {
  v8bf lo = *(const v8bf*)(p);
  v8bf hi = *(const v8bf*)(p + 16);
  v16bf r;
#pragma unroll
  for (int e = 0; e < 8; ++e) { r[e] = lo[e]; r[8 + e] = hi[e]; }
  return r;
}

// ---------------- unified WMMA bf16 GEMM ----------------
// C = act(alpha * A@B^T + bias) where A: [M,K] bf16 (lda), B: [N,K] bf16 (ldb)
// (all weights are pre-transposed to [N,K]; activations produced K-contiguous)
// Outputs (any may be null): C f32 [M,N] (ldc), C2 bf16 [M,N] (ldc2),
// CT bf16 head-transposed [B,NH,DH,S] (ctmode==1, used for the V projection).
// Async double-buffered LDS pipeline: tiles DMA'd with
// global_load_async_to_lds_b128 while 8 WMMAs run on the other buffer.
#define BMT 128
#define BNT 128
#define BKT 32
#define LDT 40   // 40 bf16 = 80B row stride -> 16B aligned chunks + frag reads

template <int ACT>
__global__ __launch_bounds__(256) void gemm_wmma(
    const bf16* __restrict__ A, int lda, long sA1, long sA2,
    const bf16* __restrict__ Bm, int ldb, long sB1, long sB2,
    float* __restrict__ C, int ldc, long sC1, long sC2,
    bf16* __restrict__ C2, int ldc2, long sC21, long sC22,
    bf16* __restrict__ CT, int ctmode,
    const float* __restrict__ bias, float alpha,
    int M, int N, int K, int inner)
{
  __shared__ __align__(16) bf16 As[2][BMT * LDT];
  __shared__ __align__(16) bf16 Bs[2][BNT * LDT];

  const int tid  = threadIdx.x;
  const int lane = tid & 31;
  const int wave = tid >> 5;
  const int hlf  = lane >> 4;    // 0 or 1
  const int idx  = lane & 15;    // 0..15
  const int wm   = wave & 1;     // 2 waves along M
  const int wn   = wave >> 1;    // 4 waves along N
  const int bM   = blockIdx.y * BMT;
  const int bN   = blockIdx.x * BNT;
  const int z    = blockIdx.z;
  const int zo   = z / inner, zi = z % inner;

  const bf16* Ab = A  + (long)zo * sA1 + (long)zi * sA2;
  const bf16* Bb = Bm + (long)zo * sB1 + (long)zi * sB2;

  v8f acc[4][2];
#pragma unroll
  for (int i = 0; i < 4; ++i)
#pragma unroll
    for (int j = 0; j < 2; ++j)
#pragma unroll
      for (int e = 0; e < 8; ++e) acc[i][j][e] = 0.0f;

  // Each tile is 128 rows x 32 bf16 = 128 x 4 16B-chunks = 512 async ops;
  // 256 threads issue 2 A-chunks + 2 B-chunks each.
  auto issue_tile = [&](int k0, int buf) {
#pragma unroll
    for (int it = 0; it < 2; ++it) {
      int c  = tid + it * 256;
      int r  = c >> 2;             // 0..127
      int kc = (c & 3) * 8;        // bf16 offset: 0,8,16,24 (16B chunks)
      const bf16* ga = Ab + (size_t)(bM + r) * lda + (k0 + kc);
      async_b128((unsigned)(size_t)&As[buf][r * LDT + kc], ga);
      int gn = bN + r;
      if (gn >= N) gn = N - 1;     // clamp: dup rows only feed discarded cols
      const bf16* gb = Bb + (size_t)gn * ldb + (k0 + kc);
      async_b128((unsigned)(size_t)&Bs[buf][r * LDT + kc], gb);
    }
  };

  issue_tile(0, 0);
  wait_async0();
  __syncthreads();

  int cur = 0;
  for (int k0 = 0; k0 < K; k0 += BKT) {
    const bool nxt = (k0 + BKT) < K;
    if (nxt) issue_tile(k0 + BKT, cur ^ 1);  // DMA next tile during compute

    v16bf afr[4], bfr[2];
#pragma unroll
    for (int i = 0; i < 4; ++i)
      afr[i] = frag_ld(&As[cur][(wm * 64 + i * 16 + idx) * LDT + hlf * 8]);
#pragma unroll
    for (int j = 0; j < 2; ++j)
      bfr[j] = frag_ld(&Bs[cur][(wn * 32 + j * 16 + idx) * LDT + hlf * 8]);

#pragma unroll
    for (int i = 0; i < 4; ++i)
#pragma unroll
      for (int j = 0; j < 2; ++j)
        acc[i][j] = __builtin_amdgcn_wmma_f32_16x16x32_bf16(
            false, afr[i], false, bfr[j], (short)0, acc[i][j], false, false);

    wait_async0();
    __syncthreads();
    cur ^= 1;
  }

  // epilogue: D layout -> VGPR r holds M = r + 8*hlf, N = idx
  float* Cb  = C  ? C  + (long)zo * sC1  + (long)zi * sC2  : nullptr;
  bf16*  C2b = C2 ? C2 + (long)zo * sC21 + (long)zi * sC22 : nullptr;
#pragma unroll
  for (int i = 0; i < 4; ++i) {
    int m0 = bM + wm * 64 + i * 16;
#pragma unroll
    for (int j = 0; j < 2; ++j) {
      int n0 = bN + wn * 32 + j * 16;
      int col = n0 + idx;
      if (col < N) {
        float bv = bias ? bias[col] : 0.0f;
#pragma unroll
        for (int r = 0; r < 8; ++r) {
          int row = m0 + r + hlf * 8;
          float v = acc[i][j][r] * alpha + bv;
          if (ACT == 1) v = 0.5f * v * (1.0f + erff(v * 0.70710678f)); // exact GELU
          if (Cb)  Cb[(size_t)row * ldc + col] = v;
          if (C2b) C2b[(size_t)row * ldc2 + col] = (bf16)v;
          if (CT && ctmode == 1) {
            // row = b*S + s ; col = h*DH + d -> CT[b][h][d][s]
            int b = row >> 9, s = row & (Sz - 1);
            int h = col >> 6, d = col & (DHz - 1);
            CT[(((size_t)b * NHz + h) * DHz + d) * Sz + s] = (bf16)v;
          }
        }
      }
    }
  }
}

// ---------------- weight convert + transpose: W[K,N] f32 -> WT[N,K] bf16 ----
__global__ __launch_bounds__(256) void transpose_cvt_k(
    const float* __restrict__ W, bf16* __restrict__ WT, int K, int N)
{
  __shared__ float t[32][33];
  const float* Wb = W + (size_t)blockIdx.z * K * N;
  bf16* WTb = WT + (size_t)blockIdx.z * N * K;
  int n0 = blockIdx.x * 32, k0 = blockIdx.y * 32;
  int tx = threadIdx.x & 31, ty = threadIdx.x >> 5;   // ty: 0..7
#pragma unroll
  for (int i = 0; i < 4; ++i) {
    int k = k0 + ty + i * 8, n = n0 + tx;
    t[ty + i * 8][tx] = (k < K && n < N) ? Wb[(size_t)k * N + n] : 0.0f;
  }
  __syncthreads();
#pragma unroll
  for (int i = 0; i < 4; ++i) {
    int n = n0 + ty + i * 8, k = k0 + tx;
    if (n < N && k < K) WTb[(size_t)n * K + k] = (bf16)t[tx][ty + i * 8];
  }
}

// ---------------- find first SEP per batch row ----------------
__global__ void sep_kernel(const int* __restrict__ ids, int* __restrict__ sep) {
  int b = threadIdx.x;
  if (b < Bz) {
    int s = Sz - 1;
    for (int i = 0; i < Sz; ++i) {
      if (ids[b * Sz + i] == SEPz) { s = i; break; }
    }
    sep[b] = s;
  }
}

// ---------------- embeddings + LayerNorm (f32 + bf16 copies) ----------------
__global__ __launch_bounds__(256) void embed_ln_k(
    const int* __restrict__ ids, const float* __restrict__ we,
    const float* __restrict__ pe, const float* __restrict__ te,
    const float* __restrict__ g, const float* __restrict__ bta,
    float* __restrict__ x, bf16* __restrict__ xh)
{
  int row = blockIdx.x;           // 0..Mtok-1
  int t   = row % Sz;
  int id  = ids[row];
  int tid = threadIdx.x;
  __shared__ float red[256];
  float vals[3]; float s = 0.0f;
#pragma unroll
  for (int i = 0; i < 3; ++i) {
    int c = tid + i * 256;
    float v = we[(size_t)id * Hz + c] + pe[(size_t)t * Hz + c] + te[c];
    vals[i] = v; s += v;
  }
  red[tid] = s; __syncthreads();
  for (int st = 128; st > 0; st >>= 1) { if (tid < st) red[tid] += red[tid + st]; __syncthreads(); }
  float mean = red[0] / Hz; __syncthreads();
  float s2 = 0.0f;
#pragma unroll
  for (int i = 0; i < 3; ++i) { float d = vals[i] - mean; s2 += d * d; }
  red[tid] = s2; __syncthreads();
  for (int st = 128; st > 0; st >>= 1) { if (tid < st) red[tid] += red[tid + st]; __syncthreads(); }
  float inv = rsqrtf(red[0] / Hz + 1e-12f);
#pragma unroll
  for (int i = 0; i < 3; ++i) {
    int c = tid + i * 256;
    float o = (vals[i] - mean) * inv * g[c] + bta[c];
    x[(size_t)row * Hz + c]  = o;
    xh[(size_t)row * Hz + c] = (bf16)o;
  }
}

// ---------------- residual add + LayerNorm (writes f32 + bf16) ----------------
__global__ __launch_bounds__(256) void add_ln_k(
    const float* __restrict__ a, const float* __restrict__ rsd,
    const float* __restrict__ g, const float* __restrict__ bta,
    float* __restrict__ out, bf16* __restrict__ outh)
{
  int row = blockIdx.x;
  int tid = threadIdx.x;
  __shared__ float red[256];
  float vals[3]; float s = 0.0f;
#pragma unroll
  for (int i = 0; i < 3; ++i) {
    int c = tid + i * 256;
    float v = a[(size_t)row * Hz + c] + rsd[(size_t)row * Hz + c];
    vals[i] = v; s += v;
  }
  red[tid] = s; __syncthreads();
  for (int st = 128; st > 0; st >>= 1) { if (tid < st) red[tid] += red[tid + st]; __syncthreads(); }
  float mean = red[0] / Hz; __syncthreads();
  float s2 = 0.0f;
#pragma unroll
  for (int i = 0; i < 3; ++i) { float d = vals[i] - mean; s2 += d * d; }
  red[tid] = s2; __syncthreads();
  for (int st = 128; st > 0; st >>= 1) { if (tid < st) red[tid] += red[tid + st]; __syncthreads(); }
  float inv = rsqrtf(red[0] / Hz + 1e-12f);
#pragma unroll
  for (int i = 0; i < 3; ++i) {
    int c = tid + i * 256;
    float o = (vals[i] - mean) * inv * g[c] + bta[c];
    out[(size_t)row * Hz + c]  = o;
    outh[(size_t)row * Hz + c] = (bf16)o;
  }
}

// --------- UniLM prefix-causal masked softmax: f32 scores -> bf16 probs ------
__global__ __launch_bounds__(256) void softmax_mask_k(
    const float* __restrict__ scores, const int* __restrict__ sep,
    bf16* __restrict__ probs)
{
  int gid = blockIdx.x;           // b*NH*S rows
  int q   = gid % Sz;
  int bh  = gid / Sz;
  int b   = bh / NHz;
  int sb  = sep[b];
  const float* rowp = scores + (size_t)bh * Sz * Sz + (size_t)q * Sz;
  bf16* ph = probs + (size_t)bh * Sz * Sz + (size_t)q * Sz;
  int tid = threadIdx.x;
  __shared__ float red[256];
  int c0 = tid, c1 = tid + 256;
  bool a0 = (c0 <= q) || (q <= sb && c0 <= sb);
  bool a1 = (c1 <= q) || (q <= sb && c1 <= sb);
  float v0 = rowp[c0] + (a0 ? 0.0f : -1e4f);
  float v1 = rowp[c1] + (a1 ? 0.0f : -1e4f);
  red[tid] = fmaxf(v0, v1); __syncthreads();
  for (int st = 128; st > 0; st >>= 1) { if (tid < st) red[tid] = fmaxf(red[tid], red[tid + st]); __syncthreads(); }
  float mx = red[0]; __syncthreads();
  float e0 = expf(v0 - mx), e1 = expf(v1 - mx);
  red[tid] = e0 + e1; __syncthreads();
  for (int st = 128; st > 0; st >>= 1) { if (tid < st) red[tid] += red[tid + st]; __syncthreads(); }
  float inv = 1.0f / red[0];
  ph[c0] = (bf16)(e0 * inv);
  ph[c1] = (bf16)(e1 * inv);
}

// ---------------- per-row cross-entropy over V ----------------
__global__ __launch_bounds__(256) void ce_row_k(
    const float* __restrict__ logits, const int* __restrict__ labels,
    float* __restrict__ nll, float* __restrict__ vld)
{
  int row = blockIdx.x;
  int tid = threadIdx.x;
  const float* lr = logits + (size_t)row * Vz;
  __shared__ float red[256];
  float mx = -1e30f;
  for (int i = tid; i < Vz; i += 256) mx = fmaxf(mx, lr[i]);
  red[tid] = mx; __syncthreads();
  for (int st = 128; st > 0; st >>= 1) { if (tid < st) red[tid] = fmaxf(red[tid], red[tid + st]); __syncthreads(); }
  mx = red[0]; __syncthreads();
  float sum = 0.0f;
  for (int i = tid; i < Vz; i += 256) sum += expf(lr[i] - mx);
  red[tid] = sum; __syncthreads();
  for (int st = 128; st > 0; st >>= 1) { if (tid < st) red[tid] += red[tid + st]; __syncthreads(); }
  if (tid == 0) {
    int lab = labels[row];
    bool valid = (lab >= 0);
    float l = valid ? (logf(red[0]) + mx - lr[lab]) : 0.0f;
    nll[row] = l;
    vld[row] = valid ? 1.0f : 0.0f;
  }
}

__global__ __launch_bounds__(256) void final_reduce_k(
    const float* __restrict__ nll, const float* __restrict__ vld,
    float* __restrict__ out, int n)
{
  __shared__ float s1[256], s2[256];
  int tid = threadIdx.x;
  float a = 0.0f, b = 0.0f;
  for (int i = tid; i < n; i += 256) { a += nll[i]; b += vld[i]; }
  s1[tid] = a; s2[tid] = b; __syncthreads();
  for (int st = 128; st > 0; st >>= 1) {
    if (tid < st) { s1[tid] += s1[tid + st]; s2[tid] += s2[tid + st]; }
    __syncthreads();
  }
  if (tid == 0) out[0] = s1[0] / fmaxf(s2[0], 1.0f);
}

// ---------------- host-side GEMM dispatch ----------------
static inline void gemm(hipStream_t s,
                        const bf16* A, int lda, long sA1, long sA2,
                        const bf16* Bm, int ldb, long sB1, long sB2,
                        float* C, int ldc, long sC1, long sC2,
                        bf16* C2, int ldc2, long sC21, long sC22,
                        bf16* CT, int ctmode,
                        const float* bias, float alpha,
                        int M, int N, int K, int batches, int inner, int act)
{
  dim3 g((N + BNT - 1) / BNT, (M + BMT - 1) / BMT, batches), blk(256);
  if (act == 1)
    gemm_wmma<1><<<g, blk, 0, s>>>(A, lda, sA1, sA2, Bm, ldb, sB1, sB2,
                                   C, ldc, sC1, sC2, C2, ldc2, sC21, sC22,
                                   CT, ctmode, bias, alpha, M, N, K, inner);
  else
    gemm_wmma<0><<<g, blk, 0, s>>>(A, lda, sA1, sA2, Bm, ldb, sB1, sB2,
                                   C, ldc, sC1, sC2, C2, ldc2, sC21, sC22,
                                   CT, ctmode, bias, alpha, M, N, K, inner);
}

extern "C" void kernel_launch(void* const* d_in, const int* in_sizes, int n_in,
                              void* d_out, int out_size, void* d_ws, size_t ws_size,
                              hipStream_t stream) {
  (void)in_sizes; (void)n_in; (void)out_size; (void)ws_size;

  const int*   input_ids = (const int*)d_in[0];
  const int*   labels    = (const int*)d_in[1];
  const float* word_emb  = (const float*)d_in[2];
  const float* pos_emb   = (const float*)d_in[3];
  const float* type_emb  = (const float*)d_in[4];
  const float* emb_g     = (const float*)d_in[5];
  const float* emb_b     = (const float*)d_in[6];
  const float* Wq = (const float*)d_in[7];  const float* bq = (const float*)d_in[8];
  const float* Wk = (const float*)d_in[9];  const float* bk = (const float*)d_in[10];
  const float* Wv = (const float*)d_in[11]; const float* bv = (const float*)d_in[12];
  const float* Wo = (const float*)d_in[13]; const float* bo = (const float*)d_in[14];
  const float* ln1g = (const float*)d_in[15]; const float* ln1b = (const float*)d_in[16];
  const float* Wi = (const float*)d_in[17]; const float* bi = (const float*)d_in[18];
  const float* Wd = (const float*)d_in[19]; const float* bd = (const float*)d_in[20];
  const float* ln2g = (const float*)d_in[21]; const float* ln2b = (const float*)d_in[22];
  const float* Wc = (const float*)d_in[23]; const float* bc = (const float*)d_in[24];

  // -------- workspace layout --------
  float* ws = (float*)d_ws;
  size_t o = 0;
  float* x      = ws + o; o += (size_t)Mtok * Hz;          // f32 activations
  float* tmp    = ws + o; o += (size_t)Mtok * Hz;          // f32 pre-LN proj
  float* scores = ws + o; o += (size_t)Bz * NHz * Sz * Sz; // f32 attn scores
  float* logits = ws + o; o += (size_t)Mtok * Vz;          // f32 logits
  float* rownll = ws + o; o += Mtok;
  float* rowvld = ws + o; o += Mtok;
  int*   sep    = (int*)(ws + o); o += Bz;

  bf16* wsb = (bf16*)(ws + o);
  size_t ob = 0;
  bf16* xh    = wsb + ob; ob += (size_t)Mtok * Hz;
  bf16* qh    = wsb + ob; ob += (size_t)Mtok * Hz;
  bf16* kh    = wsb + ob; ob += (size_t)Mtok * Hz;
  bf16* vT    = wsb + ob; ob += (size_t)Bz * NHz * DHz * Sz;  // [B,NH,DH,S]
  bf16* ctxh  = wsb + ob; ob += (size_t)Mtok * Hz;
  bf16* hffh  = wsb + ob; ob += (size_t)Mtok * FFz;
  bf16* probs = wsb + ob; ob += (size_t)Bz * NHz * Sz * Sz;
  bf16* WqT   = wsb + ob; ob += (size_t)Lz * Hz * Hz;          // [N=H, K=H]
  bf16* WkT   = wsb + ob; ob += (size_t)Lz * Hz * Hz;
  bf16* WvT   = wsb + ob; ob += (size_t)Lz * Hz * Hz;
  bf16* WoT   = wsb + ob; ob += (size_t)Lz * Hz * Hz;
  bf16* WiT   = wsb + ob; ob += (size_t)Lz * FFz * Hz;         // [N=FF, K=H]
  bf16* WdT   = wsb + ob; ob += (size_t)Lz * Hz * FFz;         // [N=H, K=FF]
  bf16* WcT   = wsb + ob; ob += (size_t)Vz * Hz;               // [N=V, K=H]

  const float scale = 0.125f;  // 1/sqrt(DH=64)

  sep_kernel<<<1, 32, 0, stream>>>(input_ids, sep);
  embed_ln_k<<<Mtok, 256, 0, stream>>>(input_ids, word_emb, pos_emb, type_emb,
                                       emb_g, emb_b, x, xh);

  // one-time per launch: weights f32 [K,N] -> bf16 [N,K]
  {
    dim3 blk(256);
    dim3 ghh((Hz + 31) / 32, (Hz + 31) / 32, Lz);
    transpose_cvt_k<<<ghh, blk, 0, stream>>>(Wq, WqT, Hz, Hz);
    transpose_cvt_k<<<ghh, blk, 0, stream>>>(Wk, WkT, Hz, Hz);
    transpose_cvt_k<<<ghh, blk, 0, stream>>>(Wv, WvT, Hz, Hz);
    transpose_cvt_k<<<ghh, blk, 0, stream>>>(Wo, WoT, Hz, Hz);
    dim3 ghf((FFz + 31) / 32, (Hz + 31) / 32, Lz);
    transpose_cvt_k<<<ghf, blk, 0, stream>>>(Wi, WiT, Hz, FFz);
    dim3 gfh((Hz + 31) / 32, (FFz + 31) / 32, Lz);
    transpose_cvt_k<<<gfh, blk, 0, stream>>>(Wd, WdT, FFz, Hz);
    dim3 ghv((Vz + 31) / 32, (Hz + 31) / 32, 1);
    transpose_cvt_k<<<ghv, blk, 0, stream>>>(Wc, WcT, Hz, Vz);
  }

  for (int l = 0; l < Lz; ++l) {
    const bf16* wqT = WqT + (size_t)l * Hz * Hz;
    const bf16* wkT = WkT + (size_t)l * Hz * Hz;
    const bf16* wvT = WvT + (size_t)l * Hz * Hz;
    const bf16* woT = WoT + (size_t)l * Hz * Hz;
    const bf16* wiT = WiT + (size_t)l * FFz * Hz;
    const bf16* wdT = WdT + (size_t)l * Hz * FFz;

    // Q, K projections -> bf16 only; V projection -> per-head transposed bf16
    gemm(stream, xh, Hz, 0, 0, wqT, Hz, 0, 0,
         nullptr, 0, 0, 0, qh, Hz, 0, 0, nullptr, 0,
         bq + l * Hz, 1.0f, Mtok, Hz, Hz, 1, 1, 0);
    gemm(stream, xh, Hz, 0, 0, wkT, Hz, 0, 0,
         nullptr, 0, 0, 0, kh, Hz, 0, 0, nullptr, 0,
         bk + l * Hz, 1.0f, Mtok, Hz, Hz, 1, 1, 0);
    gemm(stream, xh, Hz, 0, 0, wvT, Hz, 0, 0,
         nullptr, 0, 0, 0, nullptr, 0, 0, 0, vT, 1,
         bv + l * Hz, 1.0f, Mtok, Hz, Hz, 1, 1, 0);

    // scores[b,h] = scale * Q_bh @ K_bh^T  (A=[S,DH] k-contig, B=[S,DH] k-contig)
    gemm(stream,
         qh, Hz, (long)Sz * Hz, (long)DHz,
         kh, Hz, (long)Sz * Hz, (long)DHz,
         scores, Sz, (long)NHz * Sz * Sz, (long)Sz * Sz,
         nullptr, 0, 0, 0, nullptr, 0,
         nullptr, scale, Sz, Sz, DHz, Bz * NHz, NHz, 0);

    softmax_mask_k<<<Bz * NHz * Sz, 256, 0, stream>>>(scores, sep, probs);

    // ctx[b,h] = P_bh @ V_bh  (A = probs [S,S], B = vT [DH,S] k-contig)
    gemm(stream,
         probs, Sz, (long)NHz * Sz * Sz, (long)Sz * Sz,
         vT, Sz, (long)NHz * DHz * Sz, (long)DHz * Sz,
         nullptr, 0, 0, 0,
         ctxh, Hz, (long)Sz * Hz, (long)DHz, nullptr, 0,
         nullptr, 1.0f, Sz, DHz, Sz, Bz * NHz, NHz, 0);

    // output projection (f32 for residual) + residual LN
    gemm(stream, ctxh, Hz, 0, 0, woT, Hz, 0, 0,
         tmp, Hz, 0, 0, nullptr, 0, 0, 0, nullptr, 0,
         bo + l * Hz, 1.0f, Mtok, Hz, Hz, 1, 1, 0);
    add_ln_k<<<Mtok, 256, 0, stream>>>(x, tmp, ln1g + l * Hz, ln1b + l * Hz, x, xh);

    // FFN: gelu(x@Wi+bi) -> bf16 ; @Wd + bd -> f32 ; residual LN
    gemm(stream, xh, Hz, 0, 0, wiT, Hz, 0, 0,
         nullptr, 0, 0, 0, hffh, FFz, 0, 0, nullptr, 0,
         bi + (size_t)l * FFz, 1.0f, Mtok, FFz, Hz, 1, 1, 1);
    gemm(stream, hffh, FFz, 0, 0, wdT, FFz, 0, 0,
         tmp, Hz, 0, 0, nullptr, 0, 0, 0, nullptr, 0,
         bd + l * Hz, 1.0f, Mtok, Hz, FFz, 1, 1, 0);
    add_ln_k<<<Mtok, 256, 0, stream>>>(x, tmp, ln2g + l * Hz, ln2b + l * Hz, x, xh);
  }

  // logits = x @ Wc + bc : [4096,768] @ [768,21128]
  gemm(stream, xh, Hz, 0, 0, WcT, Hz, 0, 0,
       logits, Vz, 0, 0, nullptr, 0, 0, 0, nullptr, 0,
       bc, 1.0f, Mtok, Vz, Hz, 1, 1, 0);

  ce_row_k<<<Mtok, 256, 0, stream>>>(logits, labels, rownll, rowvld);
  final_reduce_k<<<1, 256, 0, stream>>>(rownll, rowvld, (float*)d_out, Mtok);
}
